// SpeMamba_17325898072382
// MI455X (gfx1250) — matrix operand order repaired and checked by reading, not compiled, not run on hardware
//
#include <hip/hip_runtime.h>
#include <hip/hip_bf16.h>

// ---------------------------------------------------------------------------
// Fused per-pixel Mamba (SpeMamba) for MI455X / gfx1250, wave32 + WMMA f16.
//   kernel 1: in_proj (WMMA) -> depthwise conv + SiLU (lane-local)
//             -> x_proj (WMMA) -> selective scan (lane-local, token-outer with
//                hoisted B/C broadcasts) -> gate silu(z) -> out_proj (WMMA)
//             -> y_pre + groupnorm partial sums
//   kernel 2: groupnorm normalize + SiLU (elementwise)
// ---------------------------------------------------------------------------

typedef __attribute__((ext_vector_type(16))) _Float16 v16h;
typedef __attribute__((ext_vector_type(8)))  float    v8f;

#define HW        4096          // 64*64
#define BSTRIDE   819200        // 200*4096
#define NPIX      8192          // 2*64*64
#define PIX_BLK   16            // pixels per workgroup
#define TOK_BLK   128           // 16 pixels * 8 tokens

__device__ __forceinline__ float sigf(float x){ return 1.f/(1.f+__expf(-x)); }
__device__ __forceinline__ float softplusf(float x){ return (x>20.f)? x : __logf(1.f+__expf(x)); }

union AFrag { v16h v; unsigned u[8]; };

// A-fragment (16x32 f16): lane holds row (lane&15); lo lanes K{0-7,16-23}, hi K{8-15,24-31}
__device__ __forceinline__ v16h load_a_frag(const _Float16* lds, int row_base, int lane,
                                            int koff, int rowstride){
  const _Float16* rp = lds + (row_base + (lane & 15)) * rowstride + koff;
  const int h8 = (lane & 16) ? 8 : 0;
  AFrag f;
  f.u[0] = *(const unsigned*)(rp + h8 + 0);
  f.u[1] = *(const unsigned*)(rp + h8 + 2);
  f.u[2] = *(const unsigned*)(rp + h8 + 4);
  f.u[3] = *(const unsigned*)(rp + h8 + 6);
  f.u[4] = *(const unsigned*)(rp + 16 + h8 + 0);
  f.u[5] = *(const unsigned*)(rp + 16 + h8 + 2);
  f.u[6] = *(const unsigned*)(rp + 16 + h8 + 4);
  f.u[7] = *(const unsigned*)(rp + 16 + h8 + 6);
  return f.v;
}

// B-fragment (32x16 f16): column n = (lane&15); lo lanes K{0-7,16-23}, hi K{8-15,24-31}.
// ldsT is K-major per column: ldsT[n][kpad]
__device__ __forceinline__ v16h load_b_frag(const _Float16* ldsT, int ncol_base, int lane,
                                            int koff, int kpad){
  const _Float16* np = ldsT + (ncol_base + (lane & 15)) * kpad + koff + ((lane & 16) ? 8 : 0);
  AFrag f;
  #pragma unroll
  for (int v = 0; v < 8; ++v) f.u[v] = *(const unsigned*)(np + ((v & 4) ? 16 : 0) + 2*(v & 3));
  return f.v;
}

__device__ __forceinline__ v8f wmma16(v16h a, v16h b, v8f c){
  return __builtin_amdgcn_wmma_f32_16x16x32_f16(false, a, false, b, (short)0, c, false, false);
}

__device__ __forceinline__ v8f vzero8(){
  v8f z;
  #pragma unroll
  for (int i = 0; i < 8; ++i) z[i] = 0.f;
  return z;
}

__global__ void init_sums_kernel(float* gsums){
  if (threadIdx.x < 16) gsums[threadIdx.x] = 0.f;
}

__global__ __launch_bounds__(256)
void mamba_fused_kernel(const float* __restrict__ x, const float* __restrict__ pe,
                        const float* __restrict__ in_proj_w, const float* __restrict__ conv_w,
                        const float* __restrict__ conv_b, const float* __restrict__ x_proj_w,
                        const float* __restrict__ dt_proj_w, const float* __restrict__ dt_proj_b,
                        const float* __restrict__ A_log, const float* __restrict__ Dp,
                        const float* __restrict__ out_proj_w,
                        float* __restrict__ ypre, float* __restrict__ gsums)
{
  __shared__ __align__(16) unsigned char smem[48064];
  _Float16* A_lds   = (_Float16*)(smem + 0);      // [128][32] xt tokens, K-padded
  _Float16* B1t     = (_Float16*)(smem + 8192);   // [112][32] in_proj  (K-major)
  _Float16* y_lds   = (_Float16*)(smem + 0);      // [128][64] aliases A_lds+B1t (dead by then)
  _Float16* xin_lds = (_Float16*)(smem + 16384);  // [128][64]
  _Float16* B2t     = (_Float16*)(smem + 32768);  // [48][64]  x_proj (K-major)
  _Float16* B3t     = (_Float16*)(smem + 38912);  // [32][64]  out_proj (K-major)
  float* Am  = (float*)(smem + 43008);            // [50][16]  A = -exp(A_log)
  float* cw  = (float*)(smem + 46208);            // [50][4]
  float* cb  = (float*)(smem + 47008);            // [50]
  float* dtw = (float*)(smem + 47208);            // [50][2]
  float* dtb = (float*)(smem + 47608);            // [50]
  float* Dv  = (float*)(smem + 47808);            // [50]
  float* red = (float*)(smem + 48008);            // [4 groups][sum,sumsq]

  const int tid  = threadIdx.x;
  const int lane = tid & 31;
  const int wave = tid >> 5;                // 0..7, one 16-token M-tile each
  const int pixbase = blockIdx.x * PIX_BLK; // 16 pixels, never crosses batch (4096 % 16 == 0)
  const int b   = pixbase >> 12;
  const int pix = pixbase & (HW - 1);

  // ---- zero staged regions (covers all f16 tile padding) ----
  for (int i = tid; i < 43008/2; i += 256) ((_Float16*)smem)[i] = (_Float16)0.f;
  if (tid < 8) red[tid] = 0.f;
  __syncthreads();

  // ---- stage weights into fragment-ready LDS tiles ----
  for (int i = tid; i < 112*32; i += 256){ int n=i>>5, k=i&31; if (n<100 && k<25) B1t[i]=(_Float16)in_proj_w[n*25+k]; }
  for (int i = tid; i < 48*64;  i += 256){ int n=i>>6, k=i&63; if (n<34  && k<50) B2t[i]=(_Float16)x_proj_w[n*50+k]; }
  for (int i = tid; i < 32*64;  i += 256){ int n=i>>6, k=i&63; if (n<25  && k<50) B3t[i]=(_Float16)out_proj_w[n*50+k]; }
  for (int i = tid; i < 800; i += 256) Am[i] = -__expf(A_log[i]);
  for (int i = tid; i < 200; i += 256) cw[i] = conv_w[i];
  if (tid < 50){ cb[tid]=conv_b[tid]; dtb[tid]=dt_proj_b[tid]; Dv[tid]=Dp[tid]; }
  if (tid < 100) dtw[tid] = dt_proj_w[tid];

  // ---- stage xt = x(+pe) as f16 tokens: token = pixel*8 + chan_group, d = c%25 ----
  for (int i = tid; i < PIX_BLK*200; i += 256){
    int c = i >> 4, p = i & 15;
    int l = c / 25, d = c - l*25;
    float v = x[b*BSTRIDE + c*HW + pix + p] + pe[d];
    A_lds[(p*8 + l)*32 + d] = (_Float16)v;
  }
  __syncthreads();

  const int tb  = wave * 16;            // token base of this wave's M-tile
  const int ln  = lane & 15;
  const int hi8 = (lane & 16) ? 8 : 0;  // C/D rows 8..15 live in high half-wave
  const int base = lane & 16;

  // ================= in_proj: [16 tok x 25] x [25 x 100] via 7 WMMA =================
  v8f acc1[7];
  #pragma unroll
  for (int t = 0; t < 7; ++t) acc1[t] = vzero8();
  {
    v16h a = load_a_frag(A_lds, tb, lane, 0, 32);
    #pragma unroll
    for (int t = 0; t < 7; ++t){
      v16h bf = load_b_frag(B1t, t*16, lane, 0, 32);
      acc1[t] = wmma16(a, bf, acc1[t]);
    }
  }
  // lane now owns feature f = 16t+ln of one sequence across 8 timesteps (regs).
  // f<50: xin channel; f>=50: z-gate channel (kept in acc1[3..6] for later).

  // ================= depthwise causal conv (k=4) + SiLU, lane-local =================
  float xcv[4][8];
  #pragma unroll
  for (int t = 0; t < 4; ++t){
    const int d = t*16 + ln;
    const bool valid = d < 50;
    const int dd = valid ? d : 0;
    const float w0=cw[dd*4+0], w1=cw[dd*4+1], w2=cw[dd*4+2], w3=cw[dd*4+3], bb=cb[dd];
    float p1=0.f, p2=0.f, p3=0.f;
    #pragma unroll
    for (int l = 0; l < 8; ++l){
      float cur = acc1[t][l];
      float v = bb + w0*p3 + w1*p2 + w2*p1 + w3*cur;
      p3 = p2; p2 = p1; p1 = cur;
      v = v * sigf(v);
      xcv[t][l] = v;
      xin_lds[(tb + hi8 + l)*64 + d] = valid ? (_Float16)v : (_Float16)0.f;
    }
  }
  __syncthreads();

  // ================= x_proj: [16 tok x 50] x [50 x 34] via 2xK32 * 3 tiles =========
  v8f accX[3];
  #pragma unroll
  for (int t = 0; t < 3; ++t) accX[t] = vzero8();
  {
    v16h a0 = load_a_frag(xin_lds, tb, lane, 0, 64);
    v16h a1 = load_a_frag(xin_lds, tb, lane, 32, 64);
    #pragma unroll
    for (int t = 0; t < 3; ++t){
      v16h b0 = load_b_frag(B2t, t*16, lane, 0, 64);
      accX[t] = wmma16(a0, b0, accX[t]);
      v16h b1 = load_b_frag(B2t, t*16, lane, 32, 64);
      accX[t] = wmma16(a1, b1, accX[t]);
    }
  }
  // x_dbl feature e = 16t+ln per lane: e<2 -> dt rank; 2..17 -> B state; 18..33 -> C state.

  // ====== hoist per-channel constants for this lane's 4 channel blocks ======
  float Arow[4][16];
  float dtw0[4], dtw1[4], bdt4[4], Dd4[4];
  #pragma unroll
  for (int t = 0; t < 4; ++t){
    const int d = t*16 + ln;
    const int dd = (d < 50) ? d : 0;
    #pragma unroll
    for (int s = 0; s < 16; ++s) Arow[t][s] = Am[dd*16 + s];
    dtw0[t] = dtw[dd*2+0]; dtw1[t] = dtw[dd*2+1];
    bdt4[t] = dtb[dd];     Dd4[t]  = Dv[dd];
  }

  // ================= selective scan, token-outer; B/C broadcast ONCE per token =====
  float h[4][16];
  #pragma unroll
  for (int t = 0; t < 4; ++t)
    #pragma unroll
    for (int s = 0; s < 16; ++s) h[t][s] = 0.f;

  // z-gate source: feature 50+d = 16*(t+3)+(2+ln) (carry to t+4 when ln>=14)
  const int zsrc = base | ((ln + 2) & 15);

  #pragma unroll
  for (int l = 0; l < 8; ++l){
    // per-token broadcasts: dt-rank scalars + B/C state rows (invariant over t)
    float r0 = __shfl(accX[0][l], base + 0, 32);
    float r1 = __shfl(accX[0][l], base + 1, 32);
    float Bv[16], Cv[16];
    #pragma unroll
    for (int s = 0; s < 16; ++s){
      const int fB = 2 + s, fC = 18 + s;
      Bv[s] = (fB < 16) ? __shfl(accX[0][l], base | (fB & 15), 32)
                        : __shfl(accX[1][l], base | (fB & 15), 32);
      Cv[s] = (fC < 32) ? __shfl(accX[1][l], base | (fC & 15), 32)
                        : __shfl(accX[2][l], base | (fC & 15), 32);
    }
    #pragma unroll
    for (int t = 0; t < 4; ++t){
      const int d = t*16 + ln;
      const bool valid = d < 50;
      float dt  = softplusf(r0*dtw0[t] + r1*dtw1[t] + bdt4[t]);
      float xv  = xcv[t][l];
      float dbx = dt * xv;
      float ysum = 0.f;
      #pragma unroll
      for (int s = 0; s < 16; ++s){
        float dA = __expf(dt * Arow[t][s]);
        h[t][s] = dA * h[t][s] + dbx * Bv[s];
        ysum += h[t][s] * Cv[s];
      }
      float yv = ysum + Dd4[t]*xv;
      const int tz1 = (t + 4 > 6) ? 6 : (t + 4);
      float za = __shfl(acc1[t+3][l], zsrc, 32);
      float zb = __shfl(acc1[tz1][l], zsrc, 32);
      float zv = (ln <= 13) ? za : zb;
      yv = yv * (zv * sigf(zv));
      y_lds[(tb + hi8 + l)*64 + d] = valid ? (_Float16)yv : (_Float16)0.f;
    }
  }
  __syncthreads();

  // ================= out_proj: [16 tok x 50] x [50 x 25] via 2xK32 * 2 tiles =======
  v8f accO[2];
  #pragma unroll
  for (int t = 0; t < 2; ++t) accO[t] = vzero8();
  {
    v16h a0 = load_a_frag(y_lds, tb, lane, 0, 64);
    v16h a1 = load_a_frag(y_lds, tb, lane, 32, 64);
    #pragma unroll
    for (int t = 0; t < 2; ++t){
      v16h b0 = load_b_frag(B3t, t*16, lane, 0, 64);
      accO[t] = wmma16(a0, b0, accO[t]);
      v16h b1 = load_b_frag(B3t, t*16, lane, 32, 64);
      accO[t] = wmma16(a1, b1, accO[t]);
    }
  }

  // ================= writeback (NHWC tokens -> NCHW) + groupnorm partials ==========
  float gs[4] = {0,0,0,0}, gq[4] = {0,0,0,0};
  #pragma unroll
  for (int nt = 0; nt < 2; ++nt){
    const int o = nt*16 + ln;
    const bool valid = o < 25;
    #pragma unroll
    for (int l = 0; l < 8; ++l){
      const int token = tb + hi8 + l;
      const int p = token >> 3;      // pixel within block
      const int lt = token & 7;      // channel-group index
      float v = accO[nt][l];
      if (valid){
        int ch = lt*25 + o;          // 0..199
        ypre[b*BSTRIDE + ch*HW + pix + p] = v;
        int g = lt >> 1;             // group = ch/50
        gs[g] += v; gq[g] += v*v;
      }
    }
  }
  #pragma unroll
  for (int g = 0; g < 4; ++g){
    atomicAdd(&red[g*2+0], gs[g]);
    atomicAdd(&red[g*2+1], gq[g]);
  }
  __syncthreads();
  if (tid < 8) atomicAdd(&gsums[b*8 + tid], red[tid]);
}

__global__ __launch_bounds__(256)
void gn_silu_kernel(const float* __restrict__ ypre, const float* __restrict__ gsums,
                    const float* __restrict__ gamma, const float* __restrict__ beta,
                    float* __restrict__ out, int n)
{
  int idx = blockIdx.x*256 + threadIdx.x;
  if (idx >= n) return;
  int b  = idx / BSTRIDE;
  int r  = idx - b*BSTRIDE;
  int ch = r >> 12;
  int g  = ch / 50;
  const float cnt = 50.f * 4096.f;
  float s = gsums[b*8 + g*2 + 0];
  float q = gsums[b*8 + g*2 + 1];
  float mu  = s / cnt;
  float var = q / cnt - mu*mu;
  float inv = rsqrtf(var + 1e-5f);
  float v  = ypre[idx];
  float xn = (v - mu) * inv * gamma[ch] + beta[ch];
  out[idx] = xn * (1.f/(1.f+__expf(-xn)));
}

extern "C" void kernel_launch(void* const* d_in, const int* in_sizes, int n_in,
                              void* d_out, int out_size, void* d_ws, size_t ws_size,
                              hipStream_t stream) {
  (void)in_sizes; (void)n_in; (void)out_size; (void)ws_size;
  const float* x         = (const float*)d_in[0];
  const float* pe        = (const float*)d_in[1];
  const float* in_proj_w = (const float*)d_in[2];
  const float* conv_w    = (const float*)d_in[3];
  const float* conv_b    = (const float*)d_in[4];
  const float* x_proj_w  = (const float*)d_in[5];
  const float* dt_proj_w = (const float*)d_in[6];
  const float* dt_proj_b = (const float*)d_in[7];
  const float* A_log     = (const float*)d_in[8];
  const float* Dp        = (const float*)d_in[9];
  const float* out_proj_w= (const float*)d_in[10];
  const float* gn_gamma  = (const float*)d_in[11];
  const float* gn_beta   = (const float*)d_in[12];

  float* gsums = (float*)d_ws;                       // 16 floats: [b][group][sum,sumsq]
  float* ypre  = (float*)((char*)d_ws + 256);        // 2*200*4096 floats

  init_sums_kernel<<<1, 32, 0, stream>>>(gsums);
  mamba_fused_kernel<<<NPIX/PIX_BLK, 256, 0, stream>>>(
      x, pe, in_proj_w, conv_w, conv_b, x_proj_w, dt_proj_w, dt_proj_b,
      A_log, Dp, out_proj_w, ypre, gsums);
  const int n = 2*200*HW;
  gn_silu_kernel<<<(n + 255)/256, 256, 0, stream>>>(ypre, gsums, gn_gamma, gn_beta,
                                                    (float*)d_out, n);
}
